// BatchGraphAttention_13305808683544
// MI455X (gfx1250) — matrix operand order repaired and checked by reading, not compiled, not run on hardware
//
#include <hip/hip_runtime.h>
#include <hip/hip_bf16.h>

typedef __attribute__((ext_vector_type(16))) _Float16 v16h;
typedef __attribute__((ext_vector_type(8)))  float    v8f;

#define Bc 8
#define Nc 2048
#define Fc 64
#define FHc 64
#define Hc 4
#define NEG_BIG (-1e10f)

// ---------------------------------------------------------------------------
// Kernel 1: feats = X @ W  per (b,h). Writes:
//   featsT f16  [B,H,FH,N]  (K-contiguous for WMMA B-operand tiles)
//   s_self f32  [B,H,N],  s_neigh f32 [B,H,N]
// Block: 256 threads, tile = 64 rows of X. Grid = B*H*(N/64) = 1024.
// ---------------------------------------------------------------------------
__global__ __launch_bounds__(256) void gat_proj(
    const float* __restrict__ X, const float* __restrict__ W,
    const float* __restrict__ a_self, const float* __restrict__ a_neigh,
    _Float16* __restrict__ featsT, float* __restrict__ sself,
    float* __restrict__ sneigh) {
  __shared__ float Xs[64][65];   // padded: conflict-free column reads
  __shared__ float Ws[64][65];
  __shared__ float red[2][64][4];

  const int bid = blockIdx.x;
  const int rt = bid & 31;          // row tile
  const int h  = (bid >> 5) & 3;
  const int b  = bid >> 7;
  const int r0 = rt * 64;
  const int t  = threadIdx.x;

  for (int q = t; q < 4096; q += 256) {
    const int i = q >> 6, f = q & 63;
    Xs[i][f] = X[((size_t)(b * Nc) + r0 + i) * Fc + f];
  }
  for (int q = t; q < 4096; q += 256) {
    const int f = q >> 6, o = q & 63;
    Ws[f][o] = W[((size_t)h * Fc + f) * FHc + o];
  }
  __syncthreads();

  const int i   = t & 63;          // row within tile (lane-contiguous)
  const int og0 = (t >> 6) * 16;   // 16-wide output column group (wave-uniform)

  float acc[16];
#pragma unroll
  for (int u = 0; u < 16; ++u) acc[u] = 0.0f;

#pragma unroll 4
  for (int f = 0; f < 64; ++f) {
    const float xv = Xs[i][f];
#pragma unroll
    for (int u = 0; u < 16; ++u) acc[u] = fmaf(xv, Ws[f][og0 + u], acc[u]);
  }

  float sp = 0.0f, np = 0.0f;
#pragma unroll
  for (int u = 0; u < 16; ++u) {
    const int o = og0 + u;
    const float v = acc[u];
    sp = fmaf(v, a_self[h * FHc + o], sp);
    np = fmaf(v, a_neigh[h * FHc + o], np);
    featsT[(((size_t)(b * Hc + h)) * FHc + o) * Nc + r0 + i] = (_Float16)v;
  }
  red[0][i][t >> 6] = sp;
  red[1][i][t >> 6] = np;
  __syncthreads();

  if (t < 64) {
    const float s = red[0][t][0] + red[0][t][1] + red[0][t][2] + red[0][t][3];
    sself[((size_t)(b * Hc + h)) * Nc + r0 + t] = s;
  } else if (t < 128) {
    const int tt = t - 64;
    const float s = red[1][tt][0] + red[1][tt][1] + red[1][tt][2] + red[1][tt][3];
    sneigh[((size_t)(b * Hc + h)) * Nc + r0 + tt] = s;
  }
}

// ---------------------------------------------------------------------------
// Kernel 2: flash-GAT, software-pipelined (double-buffered LDS).
// Block = (b, 64-row tile); 8 waves = 4 heads x 2 row subtiles.
// Each wave: 32 rows x 64 cols for one head; online softmax; P generated
// directly in WMMA A-operand layout; 8x v_wmma_f32_16x16x32_f16 per 32-chunk.
// A mask read from HBM exactly once, shared by all 4 heads via LDS.
// Chunk j+1 global loads are staged in registers while chunk j computes.
// ---------------------------------------------------------------------------
__global__ __launch_bounds__(256) void gat_flash(
    const float* __restrict__ A, const float* __restrict__ bias,
    const float* __restrict__ sself, const float* __restrict__ sneigh,
    const _Float16* __restrict__ featsT, float* __restrict__ out) {
  __shared__ float Am[2][64][33];            // A tile, padded (stride 33)
  __shared__ _Float16 fB[2][4][64][40];      // feats chunk [h][o][k], 80B rows

  const int b  = blockIdx.y;
  const int rt = blockIdx.x;              // 64-row tile index
  const int tid = threadIdx.x;
  const int w  = tid >> 5, l = tid & 31;
  const int h  = w & 3, rowsub = w >> 2;  // head, 32-row subtile
  const int lrow = l & 15, lhalf = l >> 4;
  const int kb = lhalf * 8;               // lane's K-phase in A-operand layout
  const int i0 = rt * 64 + rowsub * 32;   // wave's first global row
  const int iloc0 = rowsub * 32;          // wave's first row within LDS tile

  const float* __restrict__ sselfBH  = sself  + ((size_t)(b * Hc + h)) * Nc;
  const float* __restrict__ sneighBH = sneigh + ((size_t)(b * Hc + h)) * Nc;
  const float* __restrict__ Arow = A + (size_t)b * Nc * Nc + (size_t)(rt * 64) * Nc;

  // loader roles (block-wide cooperative staging)
  const int ar0 = tid >> 3,        ac0 = (tid & 7) * 4;        // A part 0
  const int ar1 = (tid + 256) >> 3, ac1 = ((tid + 256) & 7) * 4; // A part 1
  const int fh = tid >> 6, fo = tid & 63;                      // feats row
  const float* __restrict__ fsrc =
      (const float*)(featsT + (((size_t)(b * Hc + fh)) * FHc + fo) * Nc);

  const float ss0 = sselfBH[i0 + lrow];        // msub0 row logit
  const float ss1 = sselfBH[i0 + 16 + lrow];   // msub1 row logit

  float m0 = -1e30f, m1 = -1e30f, lsum0 = 0.0f, lsum1 = 0.0f;
  v8f acc[2][4];
#pragma unroll
  for (int mi = 0; mi < 2; ++mi)
#pragma unroll
    for (int ni = 0; ni < 4; ++ni) acc[mi][ni] = (v8f){};

  // register staging for the software pipeline
  float4 aR0, aR1, fR0, fR1, fR2, fR3;

  auto issue_loads = [&](int j0) {
    aR0 = *(const float4*)(Arow + (size_t)ar0 * Nc + j0 + ac0);
    aR1 = *(const float4*)(Arow + (size_t)ar1 * Nc + j0 + ac1);
    const float4* s4 = (const float4*)(fsrc + (j0 >> 1));  // f16 row, j0*2 bytes in
    fR0 = s4[0]; fR1 = s4[1]; fR2 = s4[2]; fR3 = s4[3];
    if (j0 + 32 < Nc)   // pull the next A chunk toward the caches early
      __builtin_prefetch(Arow + (size_t)ar0 * Nc + j0 + 32 + ac0, 0, 0);
  };

  issue_loads(0);

  for (int jc = 0; jc < Nc / 32; ++jc) {
    const int j0 = jc * 32;
    const int buf = jc & 1;

    // --- drain staged registers into LDS buffer `buf` ---
    Am[buf][ar0][ac0 + 0] = aR0.x; Am[buf][ar0][ac0 + 1] = aR0.y;
    Am[buf][ar0][ac0 + 2] = aR0.z; Am[buf][ar0][ac0 + 3] = aR0.w;
    Am[buf][ar1][ac1 + 0] = aR1.x; Am[buf][ar1][ac1 + 1] = aR1.y;
    Am[buf][ar1][ac1 + 2] = aR1.z; Am[buf][ar1][ac1 + 3] = aR1.w;
    {
      float4* d4 = (float4*)(&fB[buf][fh][fo][0]);
      d4[0] = fR0; d4[1] = fR1; d4[2] = fR2; d4[3] = fR3;
    }
    __syncthreads();

    // --- issue next chunk's global loads; latency hides under compute ---
    if (jc + 1 < Nc / 32) issue_loads(j0 + 32);

    // --- neighbor logits for this lane's 16 K-slots ---
    float sn[16];
    {
      const float4* p = (const float4*)(sneighBH + j0);
      const float4 a0 = p[lhalf * 2], a1 = p[lhalf * 2 + 1];
      const float4 b0 = p[4 + lhalf * 2], b1 = p[5 + lhalf * 2];
      sn[0] = a0.x; sn[1] = a0.y; sn[2] = a0.z; sn[3] = a0.w;
      sn[4] = a1.x; sn[5] = a1.y; sn[6] = a1.z; sn[7] = a1.w;
      sn[8] = b0.x; sn[9] = b0.y; sn[10] = b0.z; sn[11] = b0.w;
      sn[12] = b1.x; sn[13] = b1.y; sn[14] = b1.z; sn[15] = b1.w;
    }

    // --- scores (leaky + mask), tile max ---
    float p0[16], p1[16];
    float tm0 = -1e30f, tm1 = -1e30f;
#pragma unroll
    for (int e = 0; e < 16; ++e) {
      const int k = ((e < 8) ? 0 : 16) + kb + (e & 7);
      const float a0v = Am[buf][iloc0 + lrow][k];
      const float a1v = Am[buf][iloc0 + 16 + lrow][k];
      float s0 = ss0 + sn[e];
      float s1 = ss1 + sn[e];
      s0 = fmaxf(s0, 0.0f) + 0.2f * fminf(s0, 0.0f);   // leaky_relu(0.2)
      s1 = fmaxf(s1, 0.0f) + 0.2f * fminf(s1, 0.0f);
      s0 = s0 + (1.0f - a0v) * NEG_BIG;                // exact reference masking
      s1 = s1 + (1.0f - a1v) * NEG_BIG;
      p0[e] = s0; p1[e] = s1;
      tm0 = fmaxf(tm0, s0); tm1 = fmaxf(tm1, s1);
    }
    tm0 = fmaxf(tm0, __shfl_xor(tm0, 16));
    tm1 = fmaxf(tm1, __shfl_xor(tm1, 16));

    const float mn0 = fmaxf(m0, tm0), mn1 = fmaxf(m1, tm1);
    const float sc0 = __expf(m0 - mn0), sc1 = __expf(m1 - mn1);
    m0 = mn0; m1 = mn1;

    // --- exponentiate into f16 A-fragments (A-operand layout) + row sums ---
    v16h af0, af1;
    float rs0 = 0.0f, rs1 = 0.0f;
#pragma unroll
    for (int e = 0; e < 16; ++e) {
      const float e0 = __expf(p0[e] - mn0);
      const float e1 = __expf(p1[e] - mn1);
      rs0 += e0; rs1 += e1;
      af0[e] = (_Float16)e0;
      af1[e] = (_Float16)e1;
    }
    rs0 += __shfl_xor(rs0, 16);
    rs1 += __shfl_xor(rs1, 16);
    lsum0 = lsum0 * sc0 + rs0;
    lsum1 = lsum1 * sc1 + rs1;

    // --- rescale accumulators: C-layout row M = r + lhalf*8 ---
    float scv0[8], scv1[8];
#pragma unroll
    for (int r = 0; r < 8; ++r) {
      scv0[r] = __shfl(sc0, r + (lhalf << 3));
      scv1[r] = __shfl(sc1, r + (lhalf << 3));
    }
#pragma unroll
    for (int ni = 0; ni < 4; ++ni)
#pragma unroll
      for (int r = 0; r < 8; ++r) {
        acc[0][ni][r] *= scv0[r];
        acc[1][ni][r] *= scv1[r];
      }

    // --- P x V: 4 column tiles x 2 row subtiles ---
#pragma unroll
    for (int ni = 0; ni < 4; ++ni) {
      const int o = ni * 16 + lrow;
      union { v16h v; float4 f4[2]; } bu;
      const float4* bp = (const float4*)(&fB[buf][h][o][0]);
      bu.f4[0] = bp[lhalf];
      bu.f4[1] = bp[2 + lhalf];
      acc[0][ni] = __builtin_amdgcn_wmma_f32_16x16x32_f16(
          false, af0, false, bu.v, (short)0, acc[0][ni], false, false);
      acc[1][ni] = __builtin_amdgcn_wmma_f32_16x16x32_f16(
          false, af1, false, bu.v, (short)0, acc[1][ni], false, false);
    }
  }

  // --- epilogue: divide by row sum, add bias, relu, store ---
  const float il0 = 1.0f / lsum0;
  const float il1 = 1.0f / lsum1;
  float inv0[8], inv1[8];
#pragma unroll
  for (int r = 0; r < 8; ++r) {
    inv0[r] = __shfl(il0, r + (lhalf << 3));
    inv1[r] = __shfl(il1, r + (lhalf << 3));
  }
#pragma unroll
  for (int ni = 0; ni < 4; ++ni) {
    const int o = ni * 16 + lrow;
    const float bv = bias[h * FHc + o];
#pragma unroll
    for (int r = 0; r < 8; ++r) {
      const int M = r + (lhalf << 3);
      const float o0 = fmaxf(acc[0][ni][r] * inv0[r] + bv, 0.0f);
      const float o1 = fmaxf(acc[1][ni][r] * inv1[r] + bv, 0.0f);
      out[((size_t)(b * Nc) + i0 + M) * (Hc * FHc) + h * FHc + o] = o0;
      out[((size_t)(b * Nc) + i0 + 16 + M) * (Hc * FHc) + h * FHc + o] = o1;
    }
  }
}

// ---------------------------------------------------------------------------
extern "C" void kernel_launch(void* const* d_in, const int* in_sizes, int n_in,
                              void* d_out, int out_size, void* d_ws, size_t ws_size,
                              hipStream_t stream) {
  const float* X       = (const float*)d_in[0];
  const float* A       = (const float*)d_in[1];
  const float* W       = (const float*)d_in[2];
  const float* bias    = (const float*)d_in[3];
  const float* a_self  = (const float*)d_in[4];
  const float* a_neigh = (const float*)d_in[5];
  float* out = (float*)d_out;

  char* ws = (char*)d_ws;
  _Float16* featsT = (_Float16*)ws;                              // 8 MB
  float* sself  = (float*)(ws + (size_t)Bc * Hc * FHc * Nc * 2); // 256 KB
  float* sneigh = sself + (size_t)Bc * Hc * Nc;                  // 256 KB

  gat_proj<<<dim3(Bc * Hc * (Nc / 64)), 256, 0, stream>>>(
      X, W, a_self, a_neigh, featsT, sself, sneigh);
  gat_flash<<<dim3(Nc / 64, Bc), 256, 0, stream>>>(
      A, bias, sself, sneigh, featsT, out);
}
